// ST_GCNA_77893526880311
// MI455X (gfx1250) — matrix operand order repaired and checked by reading, not compiled, not run on hardware
//
#include <hip/hip_runtime.h>
#include <hip/hip_bf16.h>

// ---------------------------------------------------------------------------
// ST_GCNA fused GAT for MI455X (gfx1250, wave32, WMMA).
//  b=16384, N=22, F=H=192. Two bf16 WMMA GEMMs per 2-batch tile, all
//  intermediates kept in LDS; attention (22x22 softmax) done on VALU.
// ---------------------------------------------------------------------------

typedef __bf16 bf16x16 __attribute__((ext_vector_type(16)));
typedef __bf16 bf16x8  __attribute__((ext_vector_type(8)));
typedef float  f32x8   __attribute__((ext_vector_type(8)));
typedef unsigned short u16x4 __attribute__((ext_vector_type(4)));
typedef unsigned short u16x8 __attribute__((ext_vector_type(8)));

union BF16x16 { bf16x16 v; bf16x8 h[2]; };

#if __has_builtin(__builtin_amdgcn_global_load_async_to_lds_b32)
#define HAVE_ASYNC_COPY 1
typedef __attribute__((address_space(1))) int* as1_i32p;
typedef __attribute__((address_space(3))) int* as3_i32p;
#else
#define HAVE_ASYNC_COPY 0
#endif

__device__ __forceinline__ unsigned short f2bf(float f) {
  return __builtin_bit_cast(unsigned short, (__bf16)f);   // native RNE cvt
}
__device__ __forceinline__ float bf2f(unsigned short s) {
  return __uint_as_float(((unsigned)s) << 16);
}
__device__ __forceinline__ float elu1(float x)  { return x > 0.f ? x : (expf(x) - 1.f); }
__device__ __forceinline__ float lrelu(float x) { return x > 0.f ? x : 0.2f * x; }

__device__ __forceinline__ void wait_asynccnt0() {
#if __has_builtin(__builtin_amdgcn_s_wait_asynccnt)
  __builtin_amdgcn_s_wait_asynccnt(0);
#else
  asm volatile("s_wait_asynccnt 0x0" ::: "memory");
#endif
}

#define NEG_INF (-9e15f)

// ---------------------------------------------------------------------------
// Kernel 0: convert weights to bf16, stored COLUMN-major so WMMA B-fragments
// (lane = output column, 8-K runs at k0+8h and k0+16+8h) are 16B loads.
//   W12t[col*192 + k] = bf16( col<192 ? W1[k,col] : W2[k,col-192] )   (384x192)
//   Wot [col*384 + k] = bf16( Wo[k,col] )                             (192x384)
// ---------------------------------------------------------------------------
__global__ void stgcna_prep_weights(const float* __restrict__ W1,
                                    const float* __restrict__ W2,
                                    const float* __restrict__ Wo,
                                    unsigned short* __restrict__ W12t,
                                    unsigned short* __restrict__ Wot) {
  int idx = blockIdx.x * blockDim.x + threadIdx.x;
  if (idx < 384 * 192) {
    int col = idx / 192, k = idx - col * 192;
    float v = (col < 192) ? W1[k * 192 + col] : W2[k * 192 + (col - 192)];
    W12t[idx] = f2bf(v);
  } else if (idx < 2 * 384 * 192) {
    int j = idx - 384 * 192;
    int col = j / 384, k = j - col * 384;
    Wot[j] = f2bf(Wo[k * 192 + col]);
  }
}

// ---------------------------------------------------------------------------
// One 16x16 output tile: D(bf16,LDS) = A(bf16,LDS [M][lda]) x B(bf16,global,
// column-major [col][ldb]) accumulated in f32 over K (multiple of 32).
// A layout (16-bit A 16x32): lanes 0-15 row M=lr, K pairs 0..7 (v0-3) and
// 16..23 (v4-7); lanes 16-31 same rows, K+8 / K+24  -> two 16B reads per lane.
// B layout (16-bit B 32x16): lane%16 = column, same K order as A -> two 16B reads per lane.
// C/D layout: VGPR r: lanes0-15 -> (M=r, N=lane), lanes16-31 -> (M=r+8).
// ---------------------------------------------------------------------------
__device__ __forceinline__ void wmma_tile_bf16(
    const unsigned short* __restrict__ sA, int lda, int mbase,
    const unsigned short* __restrict__ gB, int ldb, int nbase,
    int K, unsigned short* __restrict__ sD, int ldd, int lane) {
  const int half = lane >> 4, lr = lane & 15;
  f32x8 acc = {0.f, 0.f, 0.f, 0.f, 0.f, 0.f, 0.f, 0.f};
  const unsigned short* arow = sA + (size_t)(mbase + lr) * lda + half * 8;
  const unsigned short* bcol = gB + (size_t)(nbase + lr) * ldb + half * 8;
  for (int k0 = 0; k0 < K; k0 += 32) {
    BF16x16 a, b;
    a.h[0] = *(const bf16x8*)(arow + k0);        // K = k0 + (half?8:0)  .. +7
    a.h[1] = *(const bf16x8*)(arow + k0 + 16);   // K = k0+16+(half?8:0) .. +7
    b.h[0] = *(const bf16x8*)(bcol + k0);        // K = k0 + (half?8:0)  .. +7
    b.h[1] = *(const bf16x8*)(bcol + k0 + 16);   // K = k0+16+(half?8:0) .. +7
    acc = __builtin_amdgcn_wmma_f32_16x16x32_bf16(
        false, a.v, false, b.v, (short)0, acc, false, false);
  }
  unsigned short* drow = sD + (size_t)(mbase + half * 8) * ldd + nbase + lr;
#pragma unroll
  for (int r = 0; r < 8; ++r) drow[(size_t)r * ldd] = f2bf(acc[r]);
}

// 192-length dot of bf16 (LDS) row with f32 (global) vector, vectorized loads.
__device__ __forceinline__ float dot192(const unsigned short* __restrict__ wr,
                                        const float* __restrict__ av) {
  float acc = 0.f;
#pragma unroll 2
  for (int k = 0; k < 192; k += 8) {
    u16x8 w = *(const u16x8*)(wr + k);
    float4 p = *(const float4*)(av + k);
    float4 q = *(const float4*)(av + k + 4);
    acc += bf2f(w[0]) * p.x + bf2f(w[1]) * p.y + bf2f(w[2]) * p.z + bf2f(w[3]) * p.w;
    acc += bf2f(w[4]) * q.x + bf2f(w[5]) * q.y + bf2f(w[6]) * q.z + bf2f(w[7]) * q.w;
  }
  return acc;
}

// ---------------------------------------------------------------------------
// Fused kernel: 1 workgroup (256 threads = 8 waves) handles 2 batch items,
// padded to 64 rows (rows 22..31 of each item zeroed).
// LDS map (bytes):
//   [0,      24576)  sX   bf16 [64][192]   (reused as sWho after GEMM2)
//   [24576,  73728)  sWh  bf16 [64][384]   (heads 1|2 concatenated)
//   [73728, 122880)  sHc  bf16 [64][384]   (hcat after attention+ELU)
//   [122880,124816)  sAdj f32  [22*22]
//   [124816,125840)  sE   f32  [2 head][2 kind][64]
//   [125840,133584)  sAtt f32  [2 bi][2 head][22][22]
//   [133584,134096)  sE2  f32  [2 kind][64]
//   [134096,137968)  sAtt2 f32 [2 bi][22][22]
// ---------------------------------------------------------------------------
__launch_bounds__(256, 2)
__global__ void stgcna_fused(const float* __restrict__ x,
                             const float* __restrict__ adj,
                             const float* __restrict__ a1,
                             const float* __restrict__ a2,
                             const float* __restrict__ ao,
                             const unsigned short* __restrict__ W12t,
                             const unsigned short* __restrict__ Wot,
                             float* __restrict__ out, int nbatch) {
  __shared__ __align__(16) unsigned char smem[137968];
  unsigned short* sX   = (unsigned short*)(smem);
  unsigned short* sWh  = (unsigned short*)(smem + 24576);
  unsigned short* sHc  = (unsigned short*)(smem + 73728);
  float* sAdj  = (float*)(smem + 122880);
  float* sE    = (float*)(smem + 124816);
  float* sAtt  = (float*)(smem + 125840);
  float* sE2   = (float*)(smem + 133584);
  float* sAtt2 = (float*)(smem + 134096);

  const int tid  = threadIdx.x;
  const int wave = tid >> 5;
  const int lane = tid & 31;
  const int b0   = blockIdx.x * 2;

  if (tid == 0) {  // weights are hot across all 8192 WGs -> keep them cached
    __builtin_prefetch(W12t, 0, 3);
    __builtin_prefetch(Wot, 0, 3);
  }

  // ---- adj -> LDS via async-copy path (ASYNCcnt) when available ----
#if HAVE_ASYNC_COPY
  if (tid < 484) {
    __builtin_amdgcn_global_load_async_to_lds_b32(
        (as1_i32p)(adj + tid), (as3_i32p)(sAdj + tid), 0, 0);
  }
  if (tid + 256 < 484) {
    __builtin_amdgcn_global_load_async_to_lds_b32(
        (as1_i32p)(adj + tid + 256), (as3_i32p)(sAdj + tid + 256), 0, 0);
  }
#else
  for (int i = tid; i < 22 * 22; i += 256) sAdj[i] = adj[i];
#endif

  // ---- Stage x (2 batches, zero-padded to 32 rows each), f32->bf16 ----
  for (int i4 = tid; i4 < 64 * 48; i4 += 256) {
    int row = i4 / 48, k = (i4 - row * 48) * 4;
    int bi = row >> 5, r = row & 31;
    float4 v = {0.f, 0.f, 0.f, 0.f};
    if (r < 22 && (b0 + bi) < nbatch)
      v = *(const float4*)(x + (((size_t)(b0 + bi)) * 22 + r) * 192 + k);
    u16x4 o = {f2bf(v.x), f2bf(v.y), f2bf(v.z), f2bf(v.w)};
    *(u16x4*)(sX + row * 192 + k) = o;
  }
#if HAVE_ASYNC_COPY
  wait_asynccnt0();
#endif
  __syncthreads();

  // ---- GEMM1: Wh[64,384] = X[64,192] @ W12t  (96 tiles, 6 K-steps each) ----
  for (int t = wave; t < 96; t += 8) {
    int mt = t & 3, nt = t >> 2;
    wmma_tile_bf16(sX, 192, mt * 16, W12t, 192, nt * 16, 192, sWh, 384, lane);
  }
  __syncthreads();

  // ---- attention scores: e_{src,dst}[head][row] = Wh_row . a ----
  {
    int hh = tid >> 7, kind = (tid >> 6) & 1, row = tid & 63;
    const float* av = (hh ? a2 : a1) + kind * 192;
    sE[(hh * 2 + kind) * 64 + row] = dot192(sWh + row * 384 + hh * 192, av);
  }
  __syncthreads();

  // ---- masked softmax rows: att[bi][h][i][:] (88 rows of length 22) ----
  if (tid < 88) {
    int bi = tid / 44, rest = tid % 44, h = rest / 22, i = rest % 22;
    const float* es = sE + (h * 2 + 0) * 64;
    const float* ed = sE + (h * 2 + 1) * 64;
    float ei = es[bi * 32 + i];
    float vals[22];
    float m = -3.0e38f;
#pragma unroll
    for (int j = 0; j < 22; ++j) {
      float e = lrelu(ei + ed[bi * 32 + j]);
      e = (sAdj[i * 22 + j] > 0.f) ? e : NEG_INF;
      vals[j] = e;
      m = fmaxf(m, e);
    }
    float s = 0.f;
#pragma unroll
    for (int j = 0; j < 22; ++j) { float p = expf(vals[j] - m); vals[j] = p; s += p; }
    float inv = 1.f / s;
    float* arow = sAtt + ((bi * 2 + h) * 22 + i) * 22;
#pragma unroll
    for (int j = 0; j < 22; ++j) arow[j] = vals[j] * inv;
  }
  __syncthreads();

  // ---- hcat = elu(att @ Wh) per head, 4 columns per thread-iteration ----
  for (int i4 = tid; i4 < 64 * 96; i4 += 256) {
    int row = i4 / 96, col = (i4 - row * 96) * 4;
    int bi = row >> 5, r = row & 31, h = col / 192;
    float4 v = {0.f, 0.f, 0.f, 0.f};
    if (r < 22) {
      const float* arow = sAtt + ((bi * 2 + h) * 22 + r) * 22;
      const unsigned short* wbase = sWh + (bi * 32) * 384 + col;
      float4 acc = {0.f, 0.f, 0.f, 0.f};
#pragma unroll
      for (int j = 0; j < 22; ++j) {
        u16x4 w = *(const u16x4*)(wbase + j * 384);
        float aj = arow[j];
        acc.x += aj * bf2f(w[0]);
        acc.y += aj * bf2f(w[1]);
        acc.z += aj * bf2f(w[2]);
        acc.w += aj * bf2f(w[3]);
      }
      v.x = elu1(acc.x); v.y = elu1(acc.y); v.z = elu1(acc.z); v.w = elu1(acc.w);
    }
    u16x4 o = {f2bf(v.x), f2bf(v.y), f2bf(v.z), f2bf(v.w)};
    *(u16x4*)(sHc + row * 384 + col) = o;
  }
  __syncthreads();

  // ---- GEMM2: Who[64,192] = hcat[64,384] @ Wot  (48 tiles, 12 K-steps) ----
  for (int t = wave; t < 48; t += 8) {
    int mt = t & 3, nt = t >> 2;
    wmma_tile_bf16(sHc, 384, mt * 16, Wot, 384, nt * 16, 384, sX, 192, lane);
  }
  __syncthreads();

  // ---- final scores with ao ----
  if (tid < 128) {
    int kind = tid >> 6, row = tid & 63;
    sE2[kind * 64 + row] = dot192(sX + row * 192, ao + kind * 192);  // sX = Who
  }
  __syncthreads();

  // ---- final softmax rows ----
  if (tid < 44) {
    int bi = tid / 22, i = tid % 22;
    float ei = sE2[bi * 32 + i];
    float vals[22];
    float m = -3.0e38f;
#pragma unroll
    for (int j = 0; j < 22; ++j) {
      float e = lrelu(ei + sE2[64 + bi * 32 + j]);
      e = (sAdj[i * 22 + j] > 0.f) ? e : NEG_INF;
      vals[j] = e;
      m = fmaxf(m, e);
    }
    float s = 0.f;
#pragma unroll
    for (int j = 0; j < 22; ++j) { float p = expf(vals[j] - m); vals[j] = p; s += p; }
    float inv = 1.f / s;
    float* arow = sAtt2 + (bi * 22 + i) * 22;
#pragma unroll
    for (int j = 0; j < 22; ++j) arow[j] = vals[j] * inv;
  }
  __syncthreads();

  // ---- y = elu(att2 @ Who) + x, 4 features per thread-iteration ----
  for (int i4 = tid; i4 < 2 * 22 * 48; i4 += 256) {
    int bi = i4 / (22 * 48), rem = i4 - bi * (22 * 48);
    int r = rem / 48, f = (rem - r * 48) * 4;
    if ((b0 + bi) >= nbatch) continue;
    const float* arow = sAtt2 + (bi * 22 + r) * 22;
    const unsigned short* wbase = sX + (bi * 32) * 192 + f;
    float4 acc = {0.f, 0.f, 0.f, 0.f};
#pragma unroll
    for (int j = 0; j < 22; ++j) {
      u16x4 w = *(const u16x4*)(wbase + j * 192);
      float aj = arow[j];
      acc.x += aj * bf2f(w[0]);
      acc.y += aj * bf2f(w[1]);
      acc.z += aj * bf2f(w[2]);
      acc.w += aj * bf2f(w[3]);
    }
    size_t g = (((size_t)(b0 + bi)) * 22 + r) * 192 + f;
    float4 xr = *(const float4*)(x + g);
    float4 o = {elu1(acc.x) + xr.x, elu1(acc.y) + xr.y,
                elu1(acc.z) + xr.z, elu1(acc.w) + xr.w};
    *(float4*)(out + g) = o;
  }
}

// ---------------------------------------------------------------------------
extern "C" void kernel_launch(void* const* d_in, const int* in_sizes, int n_in,
                              void* d_out, int out_size, void* d_ws, size_t ws_size,
                              hipStream_t stream) {
  const float* x   = (const float*)d_in[0];
  const float* adj = (const float*)d_in[1];
  const float* W1  = (const float*)d_in[2];
  const float* a1  = (const float*)d_in[3];
  const float* W2  = (const float*)d_in[4];
  const float* a2  = (const float*)d_in[5];
  const float* Wo  = (const float*)d_in[6];
  const float* ao  = (const float*)d_in[7];

  unsigned short* W12t = (unsigned short*)d_ws;          // 384*192 bf16
  unsigned short* Wot  = W12t + 384 * 192;               // 192*384 bf16

  stgcna_prep_weights<<<(2 * 384 * 192 + 255) / 256, 256, 0, stream>>>(
      W1, W2, Wo, W12t, Wot);

  const int nbatch = in_sizes[0] / (22 * 192);
  const int ngrp = (nbatch + 1) / 2;
  stgcna_fused<<<ngrp, 256, 0, stream>>>(x, adj, a1, a2, ao, W12t, Wot,
                                         (float*)d_out, nbatch);
}
